// Nilinker_47837345743364
// MI455X (gfx1250) — matrix operand order repaired and checked by reading, not compiled, not run on hardware
//
#include <hip/hip_runtime.h>
#include <hip/hip_bf16.h>

#define DEV_INLINE __device__ __forceinline__

typedef __attribute__((ext_vector_type(16))) __bf16 v16bf;
typedef __attribute__((ext_vector_type(8)))  float  v8f;

union Frag16 { v16bf v; uint4 q[2]; };

static constexpr int Dm    = 256;      // embedding dim
static constexpr int Vv    = 50000;    // candidate vocab
static constexpr int Bb    = 1024;     // entities
static constexpr int Cc    = 64;       // candidates per side
static constexpr int TOPK  = 10;
static constexpr int NTILES = Vv / 16;                 // 3125 (exact)
static constexpr int NT_PER_CHUNK = 32;
static constexpr int NCHUNKS = (NTILES + NT_PER_CHUNK - 1) / NT_PER_CHUNK;  // 98

// ---- workspace layout (bytes) ----
static constexpr size_t WS_CAND_BF = 0;                                   // V*D bf16
static constexpr size_t WS_WAT     = WS_CAND_BF + (size_t)Vv * Dm * 2;    // D*D bf16 (W_a^T)
static constexpr size_t WS_WCT     = WS_WAT + (size_t)Dm * Dm * 2;        // D*2D bf16 (W_c^T)
static constexpr size_t WS_ALIGN   = WS_WCT + (size_t)Dm * 2 * Dm * 2;    // 2B*D f32
static constexpr size_t WS_AGGRE   = WS_ALIGN + (size_t)2 * Bb * Dm * 4;  // 2B*D f32
static constexpr size_t WS_PHRASE  = WS_AGGRE + (size_t)2 * Bb * Dm * 4;  // B*D bf16
static constexpr size_t WS_MAXP    = WS_PHRASE + (size_t)Bb * Dm * 2;     // B*NCHUNKS f32
static constexpr size_t WS_SUMP    = WS_MAXP + (size_t)Bb * NCHUNKS * 4;  // B*NCHUNKS f32
static constexpr size_t WS_ROWMAX  = WS_SUMP + (size_t)Bb * NCHUNKS * 4;  // B f32
static constexpr size_t WS_ROWSUM  = WS_ROWMAX + (size_t)Bb * 4;          // B f32

DEV_INLINE unsigned short f2bf(float f) {
    unsigned int u = __float_as_uint(f);
    unsigned int r = u + 0x7FFFu + ((u >> 16) & 1u);   // round-to-nearest-even
    return (unsigned short)(r >> 16);
}

DEV_INLINE float negInf() { return -__builtin_inff(); }

DEV_INLINE void topk_insert(float (&bv)[TOPK], int (&bi)[TOPK], float v, int ix) {
    if (v <= bv[TOPK - 1]) return;
    bv[TOPK - 1] = v; bi[TOPK - 1] = ix;
#pragma unroll
    for (int j = TOPK - 1; j > 0; --j) {
        if (bv[j] > bv[j - 1]) {
            float tv = bv[j]; bv[j] = bv[j - 1]; bv[j - 1] = tv;
            int   ti = bi[j]; bi[j] = bi[j - 1]; bi[j - 1] = ti;
        }
    }
}

// ---------------- K0: precision/layout conversion ----------------
__global__ void k0_convert(const float* __restrict__ candE,
                           const float* __restrict__ W_a,
                           const float* __restrict__ W_c,
                           unsigned short* __restrict__ candBF,
                           unsigned short* __restrict__ WaT,
                           unsigned short* __restrict__ WcT) {
    int i = blockIdx.x * 256 + threadIdx.x;
    const int NC = Vv * Dm;
    if (i < NC) { candBF[i] = f2bf(candE[i]); return; }
    i -= NC;
    if (i < Dm * Dm) {                       // W_a^T[n][k] = W_a[k][n]
        int n = i >> 8, k = i & 255;
        WaT[i] = f2bf(W_a[k * Dm + n]);
        return;
    }
    i -= Dm * Dm;
    if (i < 2 * Dm * Dm) {                   // W_c^T[n][k] = W_c[k][n], k<512
        int n = i >> 9, k = i & 511;
        WcT[i] = f2bf(W_c[k * Dm + n]);
    }
}

// ---------------- K1a: align = tanh(words @ W_a + b_a), rows: [wl; wr] ----------------
__global__ void k1a_align(const int* __restrict__ wl_ids, const int* __restrict__ wr_ids,
                          const float* __restrict__ word_embeds,
                          const unsigned short* __restrict__ WaT,
                          const float* __restrict__ b_a,
                          float* __restrict__ alignBuf) {
    __shared__ unsigned short As[16 * 256];
    const int tid = threadIdx.x;
    const int b0 = blockIdx.x * 16;                 // global row in [0,2048)
    {   // gather + bf16-convert 16 word rows into LDS
        int row = tid >> 4;
        int colb = (tid & 15) * 16;
        int gr = b0 + row;
        int id = (gr < Bb) ? wl_ids[gr] : wr_ids[gr - Bb];
        const float* src = word_embeds + (size_t)id * Dm + colb;
#pragma unroll
        for (int i = 0; i < 16; ++i) As[row * 256 + colb + i] = f2bf(src[i]);
    }
    __syncthreads();
    const int lane = tid & 31, w = tid >> 5;
    const int hi = (lane & 16) ? 1 : 0;
    const v8f zero = {0.f,0.f,0.f,0.f,0.f,0.f,0.f,0.f};
#pragma unroll
    for (int nti = 0; nti < 2; ++nti) {
        const int n0 = (w * 2 + nti) * 16;
        const int col = n0 + (lane & 15);
        v8f acc = zero;
        for (int kk = 0; kk < 8; ++kk) {
            const int k0 = kk * 32;
            Frag16 a, bf;
            const unsigned short* ap = As + (lane & 15) * 256 + k0 + hi * 8;
            a.q[0] = *(const uint4*)(ap);
            a.q[1] = *(const uint4*)(ap + 16);
            const unsigned short* bp = WaT + (size_t)col * 256 + k0 + hi * 8;
            bf.q[0] = *(const uint4*)(bp);
            bf.q[1] = *(const uint4*)(bp + 16);
            acc = __builtin_amdgcn_wmma_f32_16x16x32_bf16(false, a.v, false, bf.v,
                                                          (short)0, acc, false, false);
        }
        const float bias = b_a[col];
#pragma unroll
        for (int j = 0; j < 8; ++j) {
            int r = j + hi * 8;
            alignBuf[(size_t)(b0 + r) * Dm + col] = tanhf(acc[j] + bias);
        }
    }
}

// ---------------- K1b: attention (scores, softmax, weighted sum) — VALU fp32 ----------------
__global__ void k1b_attend(const int* __restrict__ cl_ids, const int* __restrict__ cr_ids,
                           const float* __restrict__ candE,
                           const float* __restrict__ alignBuf,
                           float* __restrict__ aggre) {
    const int lane = threadIdx.x & 31;
    const int w = threadIdx.x >> 5;
    const int t = blockIdx.x * 8 + w;               // task 0..2047; <1024: (wl, cr); >=1024: (wr, cl)
    const int b = t & (Bb - 1);
    const int* ids = ((t < Bb) ? cr_ids : cl_ids) + b * Cc;
    const float* arow = alignBuf + (size_t)t * Dm + lane * 8;
    const float4 av0 = ((const float4*)arow)[0];
    const float4 av1 = ((const float4*)arow)[1];

    int id0 = ids[lane], id1 = ids[lane + 32];
    float s0 = 0.f, s1 = 0.f;
    for (int c = 0; c < Cc; ++c) {
        int v = __shfl((c < 32) ? id0 : id1, c & 31, 32);
        const float4* r4 = (const float4*)(candE + (size_t)v * Dm + lane * 8);
        float4 c0 = r4[0], c1 = r4[1];
        float p = av0.x*c0.x + av0.y*c0.y + av0.z*c0.z + av0.w*c0.w
                + av1.x*c1.x + av1.y*c1.y + av1.z*c1.z + av1.w*c1.w;
        for (int off = 16; off; off >>= 1) p += __shfl_xor(p, off, 32);
        if ((c & 31) == lane) { if (c < 32) s0 = p; else s1 = p; }
    }
    float m = fmaxf(s0, s1);
    for (int off = 16; off; off >>= 1) m = fmaxf(m, __shfl_xor(m, off, 32));
    float e0 = __expf(s0 - m), e1 = __expf(s1 - m);
    float ssum = e0 + e1;
    for (int off = 16; off; off >>= 1) ssum += __shfl_xor(ssum, off, 32);
    float inv = 1.f / ssum;
    float a0 = e0 * inv, a1 = e1 * inv;

    float acc[8] = {0,0,0,0,0,0,0,0};
    for (int c = 0; c < Cc; ++c) {
        int v = __shfl((c < 32) ? id0 : id1, c & 31, 32);
        float at = __shfl((c < 32) ? a0 : a1, c & 31, 32);
        const float4* r4 = (const float4*)(candE + (size_t)v * Dm + lane * 8);
        float4 c0 = r4[0], c1 = r4[1];
        acc[0] += at*c0.x; acc[1] += at*c0.y; acc[2] += at*c0.z; acc[3] += at*c0.w;
        acc[4] += at*c1.x; acc[5] += at*c1.y; acc[6] += at*c1.z; acc[7] += at*c1.w;
    }
    float* orow = aggre + (size_t)t * Dm + lane * 8;
    ((float4*)orow)[0] = make_float4(acc[0], acc[1], acc[2], acc[3]);
    ((float4*)orow)[1] = make_float4(acc[4], acc[5], acc[6], acc[7]);
}

// ---------------- K1c: phrase = tanh(concat(wl+wr, aggL+aggR) @ W_c + b_c) -> bf16 ----------------
__global__ void k1c_phrase(const int* __restrict__ wl_ids, const int* __restrict__ wr_ids,
                           const float* __restrict__ word_embeds,
                           const float* __restrict__ aggre,
                           const unsigned short* __restrict__ WcT,
                           const float* __restrict__ b_c,
                           unsigned short* __restrict__ phraseBF) {
    __shared__ unsigned short As[16 * 512];
    const int tid = threadIdx.x;
    const int b0 = blockIdx.x * 16;
    {   // build concat rows in LDS (bf16)
        int row = tid >> 4;
        int colb = (tid & 15) * 32;
        int gr = b0 + row;
        const float* wlr = word_embeds + (size_t)wl_ids[gr] * Dm;
        const float* wrr = word_embeds + (size_t)wr_ids[gr] * Dm;
        const float* agl = aggre + (size_t)gr * Dm;
        const float* agr = aggre + (size_t)(Bb + gr) * Dm;
#pragma unroll
        for (int i = 0; i < 32; ++i) {
            int col = colb + i;
            float vv = (col < Dm) ? (wlr[col] + wrr[col]) : (agl[col - Dm] + agr[col - Dm]);
            As[row * 512 + col] = f2bf(vv);
        }
    }
    __syncthreads();
    const int lane = tid & 31, w = tid >> 5;
    const int hi = (lane & 16) ? 1 : 0;
    const v8f zero = {0.f,0.f,0.f,0.f,0.f,0.f,0.f,0.f};
#pragma unroll
    for (int nti = 0; nti < 2; ++nti) {
        const int n0 = (w * 2 + nti) * 16;
        const int col = n0 + (lane & 15);
        v8f acc = zero;
        for (int kk = 0; kk < 16; ++kk) {
            const int k0 = kk * 32;
            Frag16 a, bf;
            const unsigned short* ap = As + (lane & 15) * 512 + k0 + hi * 8;
            a.q[0] = *(const uint4*)(ap);
            a.q[1] = *(const uint4*)(ap + 16);
            const unsigned short* bp = WcT + (size_t)col * 512 + k0 + hi * 8;
            bf.q[0] = *(const uint4*)(bp);
            bf.q[1] = *(const uint4*)(bp + 16);
            acc = __builtin_amdgcn_wmma_f32_16x16x32_bf16(false, a.v, false, bf.v,
                                                          (short)0, acc, false, false);
        }
        const float bias = b_c[col];
#pragma unroll
        for (int j = 0; j < 8; ++j) {
            int r = j + hi * 8;
            phraseBF[(size_t)(b0 + r) * Dm + col] = f2bf(tanhf(acc[j] + bias));
        }
    }
}

// ---------------- K2: logits = phrase @ candE^T (bf16 WMMA) + chunk softmax partials ----------------
__global__ void k2_logits(const unsigned short* __restrict__ phraseBF,
                          const unsigned short* __restrict__ candBF,
                          float* __restrict__ yp,
                          float* __restrict__ maxp, float* __restrict__ sump) {
    __shared__ unsigned short As[16 * 256];
    __shared__ float wmax[8][16];
    __shared__ float wsum[8][16];
    __shared__ float bmax[16];
    const int tid = threadIdx.x;
    const int chunk = blockIdx.x;
    const int b0 = blockIdx.y * 16;
    {   // stage 16x256 bf16 A panel (8 KB, contiguous)
        const uint4* src = (const uint4*)(phraseBF + (size_t)b0 * Dm);
        uint4* dst = (uint4*)As;
        dst[tid] = src[tid];
        dst[tid + 256] = src[tid + 256];
    }
    __syncthreads();
    const int lane = tid & 31, w = tid >> 5;
    const int hi = (lane & 16) ? 1 : 0;
    const v8f zero = {0.f,0.f,0.f,0.f,0.f,0.f,0.f,0.f};

    Frag16 afr[8];
#pragma unroll
    for (int kk = 0; kk < 8; ++kk) {
        const unsigned short* ap = As + (lane & 15) * 256 + kk * 32 + hi * 8;
        afr[kk].q[0] = *(const uint4*)(ap);
        afr[kk].q[1] = *(const uint4*)(ap + 16);
    }

    v8f acc[4];
    int ntile[4]; bool valid[4];
#pragma unroll
    for (int t = 0; t < 4; ++t) {
        ntile[t] = chunk * NT_PER_CHUNK + w * 4 + t;
        valid[t] = (ntile[t] < NTILES);
        acc[t] = zero;
    }
#pragma unroll
    for (int t = 0; t < 4; ++t) {
        if (!valid[t]) continue;
        const unsigned short* bbase =
            candBF + (size_t)(ntile[t] * 16 + (lane & 15)) * Dm + hi * 8;
        v8f a = zero;
        for (int kk = 0; kk < 8; ++kk) {
            Frag16 bf;
            const unsigned short* bp = bbase + kk * 32;
            bf.q[0] = *(const uint4*)(bp);
            bf.q[1] = *(const uint4*)(bp + 16);
            a = __builtin_amdgcn_wmma_f32_16x16x32_bf16(false, afr[kk].v, false, bf.v,
                                                        (short)0, a, false, false);
        }
        acc[t] = a;
    }

    // per-row max over this wave's tiles, reduced across the 16-lane column group
    float m[8];
#pragma unroll
    for (int j = 0; j < 8; ++j) {
        float mm = negInf();
#pragma unroll
        for (int t = 0; t < 4; ++t) if (valid[t]) mm = fmaxf(mm, acc[t][j]);
        for (int off = 1; off < 16; off <<= 1) mm = fmaxf(mm, __shfl_xor(mm, off, 32));
        m[j] = mm;
    }
    if ((lane & 15) == 0) {
        int rb = hi * 8;
#pragma unroll
        for (int j = 0; j < 8; ++j) wmax[w][rb + j] = m[j];
    }
    __syncthreads();
    if (tid < 16) {
        float mm = negInf();
        for (int ww = 0; ww < 8; ++ww) mm = fmaxf(mm, wmax[ww][tid]);
        bmax[tid] = mm;
    }
    __syncthreads();

    // exp-sums against chunk max + store raw logits
    float s[8];
#pragma unroll
    for (int j = 0; j < 8; ++j) {
        const int r = j + hi * 8;
        const float bm = bmax[r];
        float ss = 0.f;
#pragma unroll
        for (int t = 0; t < 4; ++t) {
            if (!valid[t]) continue;
            float l = acc[t][j];
            yp[(size_t)(b0 + r) * Vv + ntile[t] * 16 + (lane & 15)] = l;
            ss += __expf(l - bm);
        }
        for (int off = 1; off < 16; off <<= 1) ss += __shfl_xor(ss, off, 32);
        s[j] = ss;
    }
    if ((lane & 15) == 0) {
        int rb = hi * 8;
#pragma unroll
        for (int j = 0; j < 8; ++j) wsum[w][rb + j] = s[j];
    }
    __syncthreads();
    if (tid < 16) {
        float ss = 0.f;
        for (int ww = 0; ww < 8; ++ww) ss += wsum[ww][tid];
        maxp[(size_t)(b0 + tid) * NCHUNKS + chunk] = bmax[tid];
        sump[(size_t)(b0 + tid) * NCHUNKS + chunk] = ss;
    }
}

// ---------------- K3: combine chunk partials per row ----------------
__global__ void k3_reduce(const float* __restrict__ maxp, const float* __restrict__ sump,
                          float* __restrict__ rowmax, float* __restrict__ rowsum) {
    int r = blockIdx.x * 256 + threadIdx.x;
    if (r >= Bb) return;
    float M = negInf();
    for (int i = 0; i < NCHUNKS; ++i) M = fmaxf(M, maxp[(size_t)r * NCHUNKS + i]);
    float S = 0.f;
    for (int i = 0; i < NCHUNKS; ++i)
        S += sump[(size_t)r * NCHUNKS + i] * __expf(maxp[(size_t)r * NCHUNKS + i] - M);
    rowmax[r] = M; rowsum[r] = S;
}

// ---------------- K4: in-place normalize + top-10 ----------------
__global__ void k4_finalize(float* __restrict__ yp,
                            const float* __restrict__ rowmax, const float* __restrict__ rowsum,
                            int* __restrict__ topk) {
    __shared__ float sv[256 * TOPK];
    __shared__ int   si[256 * TOPK];
    const int r = blockIdx.x;
    const int tid = threadIdx.x;
    const float M = rowmax[r];
    const float inv = 1.f / rowsum[r];
    float bv[TOPK]; int bi[TOPK];
#pragma unroll
    for (int j = 0; j < TOPK; ++j) { bv[j] = negInf(); bi[j] = 0; }
    float* row = yp + (size_t)r * Vv;
    for (int c = tid; c < Vv; c += 256) {
        float l = row[c];
        row[c] = __expf(l - M) * inv;
        topk_insert(bv, bi, l, c);
    }
#pragma unroll
    for (int j = 0; j < TOPK; ++j) { sv[tid * TOPK + j] = bv[j]; si[tid * TOPK + j] = bi[j]; }
    __syncthreads();
    if (tid < 32) {
#pragma unroll
        for (int j = 0; j < TOPK; ++j) { bv[j] = negInf(); bi[j] = 0; }
        for (int s = tid; s < 256; s += 32)
            for (int j = 0; j < TOPK; ++j) topk_insert(bv, bi, sv[s * TOPK + j], si[s * TOPK + j]);
#pragma unroll
        for (int j = 0; j < TOPK; ++j) { sv[tid * TOPK + j] = bv[j]; si[tid * TOPK + j] = bi[j]; }
    }
    __syncthreads();
    if (tid == 0) {
#pragma unroll
        for (int j = 0; j < TOPK; ++j) { bv[j] = negInf(); bi[j] = 0; }
        for (int s = 0; s < 32; ++s)
            for (int j = 0; j < TOPK; ++j) topk_insert(bv, bi, sv[s * TOPK + j], si[s * TOPK + j]);
#pragma unroll
        for (int j = 0; j < TOPK; ++j) topk[r * TOPK + j] = bi[j];
    }
}

// ---------------- launcher ----------------
extern "C" void kernel_launch(void* const* d_in, const int* in_sizes, int n_in,
                              void* d_out, int out_size, void* d_ws, size_t ws_size,
                              hipStream_t stream) {
    (void)in_sizes; (void)n_in; (void)out_size; (void)ws_size;
    const int*   wl_ids      = (const int*)d_in[0];
    const int*   wr_ids      = (const int*)d_in[1];
    const int*   cl_ids      = (const int*)d_in[2];
    const int*   cr_ids      = (const int*)d_in[3];
    const float* word_embeds = (const float*)d_in[4];
    const float* candE       = (const float*)d_in[5];
    const float* W_a         = (const float*)d_in[6];
    const float* b_a         = (const float*)d_in[7];
    const float* W_c         = (const float*)d_in[8];
    const float* b_c         = (const float*)d_in[9];

    char* ws = (char*)d_ws;
    unsigned short* candBF   = (unsigned short*)(ws + WS_CAND_BF);
    unsigned short* WaT      = (unsigned short*)(ws + WS_WAT);
    unsigned short* WcT      = (unsigned short*)(ws + WS_WCT);
    float* alignBuf          = (float*)(ws + WS_ALIGN);
    float* aggre             = (float*)(ws + WS_AGGRE);
    unsigned short* phraseBF = (unsigned short*)(ws + WS_PHRASE);
    float* maxp              = (float*)(ws + WS_MAXP);
    float* sump              = (float*)(ws + WS_SUMP);
    float* rowmax            = (float*)(ws + WS_ROWMAX);
    float* rowsum            = (float*)(ws + WS_ROWSUM);

    float* yp   = (float*)d_out;
    int*   topk = (int*)(yp + (size_t)Bb * Vv);

    const int totalConv = Vv * Dm + Dm * Dm + 2 * Dm * Dm;
    k0_convert<<<(totalConv + 255) / 256, 256, 0, stream>>>(candE, W_a, W_c, candBF, WaT, WcT);
    k1a_align <<<2 * Bb / 16, 256, 0, stream>>>(wl_ids, wr_ids, word_embeds, WaT, b_a, alignBuf);
    k1b_attend<<<2 * Bb / 8, 256, 0, stream>>>(cl_ids, cr_ids, candE, alignBuf, aggre);
    k1c_phrase<<<Bb / 16, 256, 0, stream>>>(wl_ids, wr_ids, word_embeds, aggre, WcT, b_c, phraseBF);
    dim3 g2(NCHUNKS, Bb / 16);
    k2_logits <<<g2, 256, 0, stream>>>(phraseBF, candBF, yp, maxp, sump);
    k3_reduce <<<(Bb + 255) / 256, 256, 0, stream>>>(maxp, sump, rowmax, rowsum);
    k4_finalize<<<Bb, 256, 0, stream>>>(yp, rowmax, rowsum, topk);
}
